// GIBnet_11278584119364
// MI455X (gfx1250) — matrix operand order repaired and checked by reading, not compiled, not run on hardware
//
#include <hip/hip_runtime.h>
#include <hip/hip_bf16.h>
#include <math.h>

#define DIN   128
#define DD    64
#define NH    3
#define HD    192        // NH*DD
#define SLOPE 0.2f
#define EPS_G 1e-5f
#define TAU_G 0.5f
#define BN_EPS 1e-3f

typedef __attribute__((ext_vector_type(16))) _Float16 v16h;
typedef __attribute__((ext_vector_type(8)))  float    v8f;

__device__ __forceinline__ float lrelu(float v)   { return v > 0.f ? v : SLOPE * v; }
__device__ __forceinline__ float sigmf(float v)   { return 1.f / (1.f + __expf(-v)); }

__device__ float atomicMaxFloat(float* addr, float val) {
    int* ai = (int*)addr;
    int old = __float_as_int(*addr);
    while (__int_as_float(old) < val) {
        int assumed = old;
        old = atomicCAS(ai, assumed, __float_as_int(val));
        if (old == assumed) break;
    }
    return __int_as_float(old);
}

// ---------------- generic init ----------------
__global__ void init_kernel(float* __restrict__ p, long n, float v) {
    long i = (long)blockIdx.x * blockDim.x + threadIdx.x;
    long s = (long)gridDim.x * blockDim.x;
    for (; i < n; i += s) p[i] = v;
}

// ---------------- WMMA GEMM:  out = epilogue(A[MxK] @ W[KxNC]) ----------------
// Compile-time K and NC: fully unrolled K accumulation chain, 32-bit addressing,
// no row guards (M is a multiple of 16 for this workload: 100000 = 6250*16).
// EPI 0: plain store
// EPI 1: gumbel-sigmoid gate (K==NC==64; A doubles as x, R is the random gate)
// EPI 2: bias + leaky_relu
template<int EPI, int K, int NC>
__global__ void wmma_gemm_kernel(const float* __restrict__ A, const float* __restrict__ W,
                                 float* __restrict__ out,
                                 const float* __restrict__ bias,
                                 const float* __restrict__ R) {
    const int lane  = threadIdx.x & 31;
    const int wave  = threadIdx.x >> 5;
    const int tileM = blockIdx.x;
    const int tileN = blockIdx.y * 4 + wave;
    if (tileN * 16 >= NC) return;
    const int g    = lane >> 4;
    const int rowA = tileM * 16 + (lane & 15);
    const int col  = tileN * 16 + (lane & 15);

    v8f c = {};
#pragma unroll
    for (int kt = 0; kt < K; kt += 32) {
        v16h a, b;
#pragma unroll
        for (int v = 0; v < 8; ++v) {
            // A fragment (16-bit 16x32 layout, ISA 7.12.2)
            const int ka = kt + ((v < 4) ? 2 * v : 16 + 2 * (v - 4)) + 8 * g;
            a[2 * v]     = (_Float16)A[rowA * K + ka];
            a[2 * v + 1] = (_Float16)A[rowA * K + ka + 1];
            // B fragment (16-bit 32x16 layout)
            const int kb = kt + ((v < 4) ? 2 * v : 16 + 2 * (v - 4)) + 8 * g;
            b[2 * v]     = (_Float16)W[kb * NC + col];
            b[2 * v + 1] = (_Float16)W[(kb + 1) * NC + col];
        }
        c = __builtin_amdgcn_wmma_f32_16x16x32_f16(false, a, false, b, (short)0, c,
                                                   false, false);
    }

#pragma unroll
    for (int v = 0; v < 8; ++v) {
        const int row = tileM * 16 + v + 8 * g;   // C/D layout: VGPR v -> M = v + 8*g
        const float acc = c[v];
        if (EPI == 0) {
            out[row * NC + col] = acc;
        } else if (EPI == 1) {
            const float pro = sigmf(acc + bias[col]);
            const float xv  = A[row * K + col];            // x (K == NC == 64)
            const float rv  = R[row * NC + col];
            const float p2  = sigmf((__logf(rv + EPS_G) - __logf(1.f - rv + EPS_G) + pro)
                                    * (1.f / TAU_G));
            out[row * NC + col] = xv * p2;
        } else {
            out[row * NC + col] = lrelu(acc + bias[col]);
        }
    }
}

// ---------------- el/er: per (node, head) dot with attn vectors ----------------
__global__ void attn_proj_kernel(const float* __restrict__ hbuf,
                                 const float* __restrict__ attn_l,
                                 const float* __restrict__ attn_r,
                                 float* __restrict__ el, float* __restrict__ er, int N) {
    int i = blockIdx.x * blockDim.x + threadIdx.x;
    if (i >= N * NH) return;
    const int n = i / NH, h = i % NH;
    const float* hp = hbuf + (long)n * HD + h * DD;
    const float* al = attn_l + h * DD;
    const float* ar = attn_r + h * DD;
    float sl = 0.f, sr = 0.f;
    for (int d = 0; d < DD; ++d) { float v = hp[d]; sl += v * al[d]; sr += v * ar[d]; }
    el[i] = sl; er[i] = sr;
}

// ---------------- edge pass 1: degrees + segment max ----------------
__global__ void edge_pass1_kernel(const int* __restrict__ src, const int* __restrict__ dst,
                                  const float* __restrict__ el, const float* __restrict__ er,
                                  float* __restrict__ dout, float* __restrict__ din,
                                  float* __restrict__ mx, int E) {
    int e = blockIdx.x * blockDim.x + threadIdx.x;
    if (e >= E) return;
    const int s = src[e], d = dst[e];
    atomicAdd(&dout[s], 1.f);
    atomicAdd(&din[d], 1.f);
#pragma unroll
    for (int h = 0; h < NH; ++h) {
        float v = lrelu(el[s * NH + h] + er[d * NH + h]);
        atomicMaxFloat(&mx[d * NH + h], v);
    }
}

// ---------------- edge pass 2: exp + segment sum ----------------
__global__ void edge_pass2_kernel(const int* __restrict__ src, const int* __restrict__ dst,
                                  const float* __restrict__ el, const float* __restrict__ er,
                                  const float* __restrict__ mx, float* __restrict__ exbuf,
                                  float* __restrict__ den, int E) {
    int e = blockIdx.x * blockDim.x + threadIdx.x;
    if (e >= E) return;
    const int s = src[e], d = dst[e];
#pragma unroll
    for (int h = 0; h < NH; ++h) {
        float v  = lrelu(el[s * NH + h] + er[d * NH + h]);
        float ex = __expf(v - mx[d * NH + h]);
        exbuf[e * NH + h] = ex;
        atomicAdd(&den[d * NH + h], ex);
    }
}

// ---------------- edge pass 3: gat[dst] += alpha * h[src]  (192 lanes/edge) ---------
__global__ void edge_aggregate_kernel(const int* __restrict__ src, const int* __restrict__ dst,
                                      const float* __restrict__ hbuf,
                                      const float* __restrict__ exbuf,
                                      const float* __restrict__ den,
                                      float* __restrict__ gat, int E) {
    const int e = blockIdx.x;
    const int t = threadIdx.x;               // 0..191 -> (head, d)
    const int s = src[e], d = dst[e];
    const int h = t >> 6;
    const float alpha = exbuf[e * NH + h] / den[d * NH + h];
    atomicAdd(&gat[(long)d * HD + t], alpha * hbuf[(long)s * HD + t]);
}

// ---------------- x = sum_h leaky(gat + b) ----------------
__global__ void reduce_heads_kernel(const float* __restrict__ gat,
                                    const float* __restrict__ b_gat,
                                    float* __restrict__ x, int N) {
    int i = blockIdx.x * blockDim.x + threadIdx.x;
    if (i >= N * DD) return;
    const int n = i / DD, d = i % DD;
    float s = 0.f;
#pragma unroll
    for (int h = 0; h < NH; ++h)
        s += lrelu(gat[(long)n * HD + h * DD + d] + b_gat[h * DD + d]);
    x[i] = s;
}

__global__ void rsqrt_kernel(float* __restrict__ p, int n) {
    int i = blockIdx.x * blockDim.x + threadIdx.x;
    if (i < n) p[i] = rsqrtf(p[i]);          // degrees >= 1 (self loops)
}

// ---------------- batchnorm (training stats) ----------------
__global__ void bn_stats_kernel(const float* __restrict__ xin, float* __restrict__ sum,
                                float* __restrict__ sumsq, int N) {
    const int d  = threadIdx.x;              // 64 threads
    const int r0 = blockIdx.x * 256;
    int rend = r0 + 256; if (rend > N) rend = N;
    float s = 0.f, ss = 0.f;
    for (int r = r0; r < rend; ++r) { float v = xin[(long)r * DD + d]; s += v; ss += v * v; }
    atomicAdd(&sum[d], s);
    atomicAdd(&sumsq[d], ss);
}

__global__ void bn_apply_kernel(float* __restrict__ xin, const float* __restrict__ sum,
                                const float* __restrict__ sumsq,
                                const float* __restrict__ gamma, const float* __restrict__ beta,
                                int N) {
    int i = blockIdx.x * blockDim.x + threadIdx.x;
    if (i >= N * DD) return;
    const int d   = i % DD;
    const float mu  = sum[d] / (float)N;
    const float var = sumsq[d] / (float)N - mu * mu;
    xin[i] = gamma[d] * (xin[i] - mu) * rsqrtf(var + BN_EPS) + beta[d];
}

// ---------------- graph conv scatter: agg[dst] += in[src]*do[src] ----------------
__global__ void gconv_scatter_kernel(const int* __restrict__ src, const int* __restrict__ dst,
                                     const float* __restrict__ xin,
                                     const float* __restrict__ dnorm,
                                     float* __restrict__ agg, int E) {
    long i = (long)blockIdx.x * blockDim.x + threadIdx.x;
    if (i >= (long)E * DD) return;
    const int e = (int)(i >> 6);
    const int d = (int)(i & 63);
    const int s = src[e], t = dst[e];
    if (d == 0) __builtin_prefetch(&src[e + 64], 0, 0);   // global_prefetch_b8
    atomicAdd(&agg[(long)t * DD + d], xin[(long)s * DD + d] * dnorm[s]);
}

__global__ void scale_rows_kernel(float* __restrict__ p, const float* __restrict__ di, int N) {
    int i = blockIdx.x * blockDim.x + threadIdx.x;
    if (i >= N * DD) return;
    p[i] *= di[i >> 6];
}

__global__ void combine3_kernel(const float* __restrict__ a, const float* __restrict__ b,
                                const float* __restrict__ c, float* __restrict__ out, long n) {
    long i = (long)blockIdx.x * blockDim.x + threadIdx.x;
    if (i < n) out[i] = (a[i] + b[i] + c[i]) * (1.f / 3.f);
}

// ---------------- row l2-normalize (one wave32 per 64-wide row) ----------------
__global__ void l2norm_kernel(const float* __restrict__ in, float* __restrict__ out, int N) {
    const int w    = (blockIdx.x * blockDim.x + threadIdx.x) >> 5;
    const int lane = threadIdx.x & 31;
    if (w >= N) return;
    float v0 = in[(long)w * DD + lane];
    float v1 = in[(long)w * DD + lane + 32];
    float sq = v0 * v0 + v1 * v1;
#pragma unroll
    for (int o = 16; o > 0; o >>= 1) sq += __shfl_xor(sq, o, 32);
    const float r = rsqrtf(fmaxf(sq, 1e-12f));
    out[(long)w * DD + lane]      = v0 * r;
    out[(long)w * DD + lane + 32] = v1 * r;
}

// =====================================================================
extern "C" void kernel_launch(void* const* d_in, const int* in_sizes, int n_in,
                              void* d_out, int out_size, void* d_ws, size_t ws_size,
                              hipStream_t stream) {
    const float* feat   = (const float*)d_in[0];
    const int*   src    = (const int*)  d_in[1];
    const int*   dst    = (const int*)  d_in[2];
    const float* r1     = (const float*)d_in[3];
    const float* r2     = (const float*)d_in[4];
    const float* W_gat  = (const float*)d_in[5];
    const float* b_gat  = (const float*)d_in[6];
    const float* attn_l = (const float*)d_in[7];
    const float* attn_r = (const float*)d_in[8];
    const float* W1     = (const float*)d_in[9];
    const float* b1     = (const float*)d_in[10];
    const float* W2     = (const float*)d_in[11];
    const float* b2     = (const float*)d_in[12];
    const float* g1     = (const float*)d_in[13];
    const float* be1    = (const float*)d_in[14];
    const float* g2     = (const float*)d_in[15];
    const float* be2    = (const float*)d_in[16];
    const float* Wl     = (const float*)d_in[17];
    const float* bl     = (const float*)d_in[18];

    const int N = in_sizes[0] / DIN;
    const int E = in_sizes[1];
    float* out = (float*)d_out;                 // [h_t | h_s | z], each N*64

    // ---- workspace carve-out (floats) ----
    float* ws    = (float*)d_ws;
    float* hbuf  = ws;                          // N*192
    float* gat   = hbuf + (long)N * HD;         // N*192
    float* xbuf  = gat + (long)N * HD;          // N*64
    float* el    = xbuf + (long)N * DD;         // N*3
    float* er    = el + (long)N * NH;
    float* mx    = er + (long)N * NH;
    float* den   = mx + (long)N * NH;
    float* dout  = den + (long)N * NH;          // N  (becomes do = deg_out^-0.5)
    float* din   = dout + N;                    // N  (becomes di = deg_in^-0.5)
    float* exbuf = din + N;                     // E*3
    float* sums  = exbuf + (long)E * NH;        // 4*64: sum1,ss1,sum2,ss2
    // regions reused once hbuf/gat are dead:
    float* in1   = gat;                         // N*64
    float* in2   = gat + (long)N * DD;          // N*64
    float* t1    = gat + 2L * N * DD;           // N*64
    float* t2    = hbuf;                        // N*64
    float* P     = hbuf + (long)N * DD;         // N*64
    float* Q     = hbuf + 2L * N * DD;          // N*64

    const int TB = 256;
    const int mtiles = (N + 15) / 16;
    auto blocks = [](long n, int tb) { return (int)((n + tb - 1) / tb); };

    // ---- init accumulators ----
    init_kernel<<<1024, TB, 0, stream>>>(dout, 2L * N, 0.f);             // dout+din
    init_kernel<<<1024, TB, 0, stream>>>(mx, (long)N * NH, -3.0e38f);
    init_kernel<<<1024, TB, 0, stream>>>(den, (long)N * NH, 0.f);
    init_kernel<<<4096, TB, 0, stream>>>(gat, (long)N * HD, 0.f);
    init_kernel<<<1, TB, 0, stream>>>(sums, 256, 0.f);

    // ---- GATConv projection: h = feat @ W_gat  (WMMA, K=128, NC=192) ----
    wmma_gemm_kernel<0, DIN, HD><<<dim3(mtiles, 3), 128, 0, stream>>>(feat, W_gat, hbuf,
                                                                      nullptr, nullptr);
    attn_proj_kernel<<<blocks((long)N * NH, TB), TB, 0, stream>>>(hbuf, attn_l, attn_r,
                                                                  el, er, N);

    // ---- edge softmax + aggregation ----
    edge_pass1_kernel<<<blocks(E, TB), TB, 0, stream>>>(src, dst, el, er, dout, din, mx, E);
    rsqrt_kernel<<<blocks(2L * N, TB), TB, 0, stream>>>(dout, 2 * N);    // do, di
    edge_pass2_kernel<<<blocks(E, TB), TB, 0, stream>>>(src, dst, el, er, mx, exbuf, den, E);
    edge_aggregate_kernel<<<E, HD, 0, stream>>>(src, dst, hbuf, exbuf, den, gat, E);
    reduce_heads_kernel<<<blocks((long)N * DD, TB), TB, 0, stream>>>(gat, b_gat, xbuf, N);

    // ---- gumbel-sigmoid gates (WMMA + fused epilogue, K=NC=64) ----
    wmma_gemm_kernel<1, DD, DD><<<dim3(mtiles, 1), 128, 0, stream>>>(xbuf, W1, in1, b1, r1);
    wmma_gemm_kernel<1, DD, DD><<<dim3(mtiles, 1), 128, 0, stream>>>(xbuf, W2, in2, b2, r2);

    // ---- batch-stat batchnorm ----
    bn_stats_kernel<<<blocks(N, 256), DD, 0, stream>>>(in1, sums, sums + 64, N);
    bn_apply_kernel<<<blocks((long)N * DD, TB), TB, 0, stream>>>(in1, sums, sums + 64, g1, be1, N);
    bn_stats_kernel<<<blocks(N, 256), DD, 0, stream>>>(in2, sums + 128, sums + 192, N);
    bn_apply_kernel<<<blocks((long)N * DD, TB), TB, 0, stream>>>(in2, sums + 128, sums + 192, g2, be2, N);

    const long ND = (long)N * DD;
    const long ED = (long)E * DD;

    // ---- teacher branch: two gconvs, combine, Wl + leaky, l2norm ----
    init_kernel<<<4096, TB, 0, stream>>>(t1, ND, 0.f);
    gconv_scatter_kernel<<<blocks(ED, TB), TB, 0, stream>>>(src, dst, in1, dout, t1, E);
    scale_rows_kernel<<<blocks(ND, TB), TB, 0, stream>>>(t1, din, N);
    init_kernel<<<4096, TB, 0, stream>>>(t2, ND, 0.f);
    gconv_scatter_kernel<<<blocks(ED, TB), TB, 0, stream>>>(src, dst, t1, dout, t2, E);
    scale_rows_kernel<<<blocks(ND, TB), TB, 0, stream>>>(t2, din, N);
    combine3_kernel<<<blocks(ND, TB), TB, 0, stream>>>(t1, t2, in1, P, ND);
    wmma_gemm_kernel<2, DD, DD><<<dim3(mtiles, 1), 128, 0, stream>>>(P, Wl, Q, bl, nullptr);
    l2norm_kernel<<<blocks((long)N * 32, TB), TB, 0, stream>>>(Q, out, N);

    // ---- student branch: two gconvs, combine, l2norm ----
    init_kernel<<<4096, TB, 0, stream>>>(t1, ND, 0.f);
    gconv_scatter_kernel<<<blocks(ED, TB), TB, 0, stream>>>(src, dst, in2, dout, t1, E);
    scale_rows_kernel<<<blocks(ND, TB), TB, 0, stream>>>(t1, din, N);
    init_kernel<<<4096, TB, 0, stream>>>(t2, ND, 0.f);
    gconv_scatter_kernel<<<blocks(ED, TB), TB, 0, stream>>>(src, dst, t1, dout, t2, E);
    scale_rows_kernel<<<blocks(ND, TB), TB, 0, stream>>>(t2, din, N);
    combine3_kernel<<<blocks(ND, TB), TB, 0, stream>>>(t1, t2, in2, P, ND);
    l2norm_kernel<<<blocks((long)N * 32, TB), TB, 0, stream>>>(P, out + ND, N);

    // ---- z = l2norm(x) ----
    l2norm_kernel<<<blocks((long)N * 32, TB), TB, 0, stream>>>(xbuf, out + 2 * ND, N);
}
